// RecurrentLayer_47957604827517
// MI455X (gfx1250) — matrix operand order repaired and checked by reading, not compiled, not run on hardware
//
#include <hip/hip_runtime.h>

// Toolchain discriminator for the TDM builtin arity (see CDNA5_HIP.md):
// the 6-arg __builtin_amdgcn_tensor_load_to_lds ships only with the
// toolchain that also has the gfx1250 TDM header.
#if __has_include(<hip/amd_detail/amd_gfx1250_TDM.h>)
#define TDM_6ARG 1
#endif

#if __has_builtin(__builtin_amdgcn_tensor_load_to_lds) && \
    __has_builtin(__builtin_amdgcn_tensor_store_from_lds)
#define USE_TDM 1
#endif

// ---------------------------------------------------------------------------
// RecurrentLayer for MI455X (gfx1250, wave32, WMMA, TDM).
//   Phase 1: x_proj = x @ W_in^T + b_in + 0.05*noise  -> written into d_out.
//   Phase 2: persistent 16-block scan kernel; W_hh slice in LDS (bf16);
//            hidden state pulled per-step global->LDS via tensor_load_to_lds
//            and pushed back LDS->global via tensor_store_from_lds (TDM both
//            directions), grid-synced with atomic counters + s_sleep spin.
// All matrix math via v_wmma_f32_16x16x32_bf16 (f32 accumulate).
// ---------------------------------------------------------------------------

typedef __attribute__((ext_vector_type(16))) __bf16 v16bf;
typedef __attribute__((ext_vector_type(8)))  float  v8f;
typedef __attribute__((ext_vector_type(4)))  unsigned int v4u;
typedef __attribute__((ext_vector_type(8)))  int v8i;
typedef __attribute__((ext_vector_type(4)))  int v4i;

union V16 {
  unsigned int   u[8];
  unsigned short s[16];
  v16bf          v;
};

constexpr int   SEQ    = 4096;
constexpr int   BATCH  = 32;
constexpr int   IN_DIM = 128;
constexpr int   HID    = 512;
constexpr float ALPHA  = 0.1f;
constexpr float NOISE_SCALE = 0.05f;
constexpr int   NBLK   = 16;          // persistent blocks in scan kernel

// Native bf16 convert: lowers to v_cvt_pk_bf16_f32 (verified in round 2).
__device__ __forceinline__ unsigned short f2bf(float f) {
  return __builtin_bit_cast(unsigned short, (__bf16)f);
}
__device__ __forceinline__ float bf2f(unsigned short h) {
  return __uint_as_float(((unsigned int)h) << 16);
}
__device__ __forceinline__ void store_bf8(unsigned short* dst, float4 a, float4 b) {
  dst[0] = f2bf(a.x); dst[1] = f2bf(a.y); dst[2] = f2bf(a.z); dst[3] = f2bf(a.w);
  dst[4] = f2bf(b.x); dst[5] = f2bf(b.y); dst[6] = f2bf(b.z); dst[7] = f2bf(b.w);
}

// --------------------------- workspace zeroing -----------------------------
__global__ void rnn_zero_ws_kernel(unsigned* __restrict__ p, int n) {
  int i = blockIdx.x * blockDim.x + threadIdx.x;
  if (i < n) p[i] = 0u;
}

// --------------------------- phase 1: projection ---------------------------
// grid (T*B/16, HID/128), block 256 (8 waves). Each wave: one 16x16 tile,
// K = 128 (4 chained WMMA).  out[r][h] = sum_k x[r][k]*W_in[h][k] + b_in[h]
//                                        + 0.05*noise[t][h]
__global__ void __launch_bounds__(256) rnn_proj_kernel(
    const float* __restrict__ x, const float* __restrict__ W_in,
    const float* __restrict__ b_in, const float* __restrict__ noise,
    float* __restrict__ out)
{
  __shared__ unsigned short xs[16 * IN_DIM];     // 4 KB bf16, swizzled granules

  const int  tid     = threadIdx.x;
  const long rowbase = (long)blockIdx.x * 16;    // flattened (t*B + b) row base

  // Stage 16 rows of x into LDS as bf16 (one 8-elem granule per thread).
  {
    const int m = tid >> 4;                      // 0..15 row
    const int g = tid & 15;                      // 0..15 granule of 8 elems
    const float* src = x + (rowbase + m) * IN_DIM + g * 8;
    float4 a = ((const float4*)src)[0];
    float4 b = ((const float4*)src)[1];
    store_bf8(&xs[m * IN_DIM + ((g ^ (m & 7)) << 3)], a, b);
  }
  __syncthreads();

  const int lane = tid & 31;
  const int wave = tid >> 5;
  const int hf   = lane >> 4;                    // lane group (0|1)
  const int ln   = lane & 15;
  const int hcol = blockIdx.y * 128 + wave * 16 + ln;   // output column h

  v8f acc = {};
#pragma unroll
  for (int kk = 0; kk < 4; ++kk) {
    V16 A, B;
    // A (16x32 bf16): lane(ln,hf) holds row ln, Ks 32kk+16hf..+15
    const int ga = 4 * kk + 2 * hf;
    *(uint4*)&A.u[0] = *(const uint4*)&xs[ln * IN_DIM + (( ga      ^ (ln & 7)) << 3)];
    *(uint4*)&A.u[4] = *(const uint4*)&xs[ln * IN_DIM + (((ga + 1) ^ (ln & 7)) << 3)];
    // B (32x16 bf16): lane(ln,hf) holds column ln, Ks 32kk+16hf..+15 (row of W_in)
    const float* wsrc = W_in + (long)hcol * IN_DIM + 32 * kk + 16 * hf;
    float4 q0 = ((const float4*)wsrc)[0];
    float4 q1 = ((const float4*)wsrc)[1];
    float4 q2 = ((const float4*)wsrc)[2];
    float4 q3 = ((const float4*)wsrc)[3];
    store_bf8(&B.s[0], q0, q1);
    store_bf8(&B.s[8], q2, q3);
    acc = __builtin_amdgcn_wmma_f32_16x16x32_bf16(
        false, A.v, false, B.v, (short)0, acc, false, false);
  }

  const int   t   = (int)(rowbase >> 5);          // 16 rows always within one t
  const float add = b_in[hcol] + NOISE_SCALE * noise[(long)t * HID + hcol];
  float* orow = out + rowbase * HID;
#pragma unroll
  for (int r = 0; r < 8; ++r)                     // C/D layout: M = r + 8*hf
    orow[(8 * hf + r) * HID + hcol] = acc[r] + add;
}

// ----------------------------- phase 2: scan -------------------------------
// 16 persistent blocks x 128 threads (4 waves, 1 per SIMD). Block `blk` owns
// hidden columns [32*blk, 32*blk+32). Per step:
//   1) TDM load: full bf16 hidden state (32x512 = 32KB) global hbuf -> LDS.
//   2) (32x512)@(512x32) via 4 waves x 16 chained bf16 WMMAs.
//   3) leak+relu epilogue; new 32x32 bf16 slice staged compactly in LDS.
//   4) TDM store: slice LDS -> global hbuf (strided 2D tile).
//   5) grid barrier via atomic counter + s_sleep spin.
__global__ void __launch_bounds__(128) rnn_scan_kernel(
    const float* __restrict__ W_hh, const float* __restrict__ b_hh,
    float* __restrict__ out, unsigned* __restrict__ cnt,
    unsigned short* __restrict__ hbuf)
{
  __shared__ unsigned short Wl[32 * HID];        // 32 KB: W_hh slice, bf16 swizzled
  __shared__ unsigned short hl[BATCH * HID];     // 32 KB: hidden state, bf16 LINEAR
  // Compact 32x32 bf16 staging for the TDM store aliases the head of hl:
  // that region is dead between the last hl read (epilogue, fenced by a
  // __syncthreads) and the next step's TDM load which rewrites all of hl.
  unsigned short* hs = hl;                       // [32 rows][32 cols] = 2 KB

  const int tid  = threadIdx.x;
  const int blk  = blockIdx.x;
  const int lane = tid & 31;
  const int wave = tid >> 5;
  const int hf   = lane >> 4;
  const int ln   = lane & 15;
  const int m0   = (wave & 1) << 4;              // batch-tile base (0|16)
  const int n0l  = (wave >> 1) << 4;             // local col-tile base (0|16)

  // Load this block's 32 rows of W_hh into LDS as bf16 (swizzled granules).
#pragma unroll 4
  for (int i = 0; i < 16; ++i) {
    const int G = tid + i * 128;                 // 0..2047 granules
    const int n = G >> 6;                        // local row 0..31
    const int g = G & 63;                        // granule within row
    const float* src = W_hh + (long)(blk * 32 + n) * HID + g * 8;
    float4 a = ((const float4*)src)[0];
    float4 b = ((const float4*)src)[1];
    store_bf8(&Wl[n * HID + ((g ^ (n & 7)) << 3)], a, b);
  }

  const int   lcol = n0l + ln;                   // local column 0..31
  const int   hcol = blk * 32 + lcol;            // this lane's global column
  const float bias = b_hh[hcol];

#if defined(USE_TDM)
  typedef __attribute__((address_space(3))) unsigned short lds_us;
  const unsigned lds_base = (unsigned)(unsigned long long)(lds_us*)hl;

  // D# LOAD: 2D tile 32 rows x 256 dwords (32x512 bf16 contiguous),
  // global hbuf -> LDS hl.
  const unsigned long long gaL = (unsigned long long)hbuf;
  v4u g0L;
  g0L.x = 1u;                                    // count=1, no gather/restore
  g0L.y = lds_base;                              // lds_addr (bytes)
  g0L.z = (unsigned)gaL;                         // global_addr[31:0]
  g0L.w = (unsigned)((gaL >> 32) & 0x01FFFFFFull) | (2u << 30);  // type=2
  v8i g1L;
  g1L[0] = (int)(2u << 16);                      // wg_mask=0, data_size=2 (4B)
  g1L[1] = (int)(256u << 16);                    // tensor_dim0[15:0]=256
  g1L[2] = (int)(32u << 16);                     // dim0 hi=0, tensor_dim1=32
  g1L[3] = (int)(256u << 16);                    // dim1 hi=0, tile_dim0=256
  g1L[4] = 32;                                   // tile_dim1=32, tile_dim2=0
  g1L[5] = 256;                                  // tensor_dim0_stride=256
  g1L[6] = 0;
  g1L[7] = 0;

  // D# STORE: 2D tile 32 rows x 16 dwords (32x32 bf16 slice), LDS hs ->
  // global hbuf + 32*blk columns, row stride 256 dwords.
  const unsigned long long gaS = (unsigned long long)(hbuf + blk * 32);
  v4u g0S;
  g0S.x = 1u;
  g0S.y = lds_base;                              // hs aliases hl's head
  g0S.z = (unsigned)gaS;
  g0S.w = (unsigned)((gaS >> 32) & 0x01FFFFFFull) | (2u << 30);
  v8i g1S;
  g1S[0] = (int)(2u << 16);                      // data_size=2 (4B)
  g1S[1] = (int)(256u << 16);                    // tensor_dim0=256
  g1S[2] = (int)(32u << 16);                     // tensor_dim1=32
  g1S[3] = (int)(16u << 16);                     // tile_dim0=16 dwords
  g1S[4] = 32;                                   // tile_dim1=32
  g1S[5] = 256;                                  // tensor_dim0_stride=256
  g1S[6] = 0;
  g1S[7] = 0;
  v4i gz = {0, 0, 0, 0};                         // groups 2/3 unused (2D)
#endif

  __syncthreads();

  for (int t = 0; t < SEQ; ++t) {
    // 1) Pull full h_{t-1} (bf16, 32KB) from global exchange buffer into LDS.
#if defined(USE_TDM)
    if (wave == 0) {                             // wave-uniform: single TDM op
#if defined(TDM_6ARG)
      v8i g4 = {0, 0, 0, 0, 0, 0, 0, 0};
      __builtin_amdgcn_tensor_load_to_lds(g0L, g1L, gz, gz, g4, /*cpol DEV*/ 16);
#else
      __builtin_amdgcn_tensor_load_to_lds(g0L, g1L, gz, gz, /*cpol DEV*/ 16);
#endif
    }
    __builtin_amdgcn_s_wait_tensorcnt(0);
#else
#pragma unroll 4
    for (int i = 0; i < 16; ++i) {
      const int G = tid + i * 128;
      const int b = G >> 6;
      const int g = G & 63;
      uint4 q = *(const uint4*)(hbuf + b * HID + g * 8);
      *(uint4*)&hl[b * HID + g * 8] = q;         // linear layout
    }
#endif
    __syncthreads();

    // Prefetch next timestep's x_proj slice while we compute (CDNA5 path).
    if (t + 1 < SEQ) {
      const float* np = out + ((long)(t + 1) * BATCH + (m0 + 8 * hf)) * HID + hcol;
      __builtin_prefetch(np, 0, 1);
    }

    // 2) WMMA chain: acc = h_{t-1} @ W_hh[slice]^T
    v8f acc = {};
    const int bm = m0 + ln;                      // A row = batch index
    const int nn = n0l + ln;                     // B col = local hidden col
#pragma unroll
    for (int kk = 0; kk < 16; ++kk) {            // K = 512 in chunks of 32
      V16 A, B;
      // A (16x32 bf16, hl LINEAR): lane(ln,hf): Ks 32kk+16hf..+15
      *(uint4*)&A.u[0] = *(const uint4*)&hl[bm * HID + 32 * kk + 16 * hf];
      *(uint4*)&A.u[4] = *(const uint4*)&hl[bm * HID + 32 * kk + 8 + 16 * hf];
      // B (32x16 bf16, Wl swizzled): lane(ln,hf): Ks 32kk+16hf..+15
      const int gb = 4 * kk + 2 * hf;
      *(uint4*)&B.u[0] = *(const uint4*)&Wl[nn * HID + (( gb      ^ (nn & 7)) << 3)];
      *(uint4*)&B.u[4] = *(const uint4*)&Wl[nn * HID + (((gb + 1) ^ (nn & 7)) << 3)];
      acc = __builtin_amdgcn_wmma_f32_16x16x32_bf16(
          false, A.v, false, B.v, (short)0, acc, false, false);
    }

    // 3) Epilogue: h_t = relu(0.9*h + 0.1*(acc + b_hh + x_proj[t])).
    //    Compute first (reads hl), then sync, then write staging/outputs.
    float* orow = out + (long)t * BATCH * HID;
    float hn[8];
#pragma unroll
    for (int r = 0; r < 8; ++r) {
      const int   b    = m0 + 8 * hf + r;        // C/D layout: M = r + 8*hf
      const float pre  = orow[b * HID + hcol];   // x_proj + b_in + noise term
      const float hold = bf2f(hl[b * HID + hcol]);
      float v = (1.0f - ALPHA) * hold + ALPHA * (acc[r] + bias + pre);
      hn[r] = fmaxf(v, 0.0f);
    }
    __syncthreads();                             // last hl reads done -> hs ok
#pragma unroll
    for (int r = 0; r < 8; ++r) {
      const int b = m0 + 8 * hf + r;
      orow[b * HID + hcol] = hn[r];              // fp32 output state
#if defined(USE_TDM)
      hs[b * 32 + lcol] = f2bf(hn[r]);           // compact bf16 slice staging
#else
      hbuf[b * HID + hcol] = f2bf(hn[r]);        // direct bf16 exchange copy
#endif
    }

    // 4) Push the new bf16 slice back to the global exchange buffer.
#if defined(USE_TDM)
    __syncthreads();                             // staging complete in LDS
    if (wave == 0) {
#if defined(TDM_6ARG)
      v8i g4 = {0, 0, 0, 0, 0, 0, 0, 0};
      __builtin_amdgcn_tensor_store_from_lds(g0S, g1S, gz, gz, g4, /*cpol DEV*/ 16);
#else
      __builtin_amdgcn_tensor_store_from_lds(g0S, g1S, gz, gz, /*cpol DEV*/ 16);
#endif
    }
    __builtin_amdgcn_s_wait_tensorcnt(0);        // slice visible before arrive
#endif

    // 5) Grid-wide barrier: release stores, atomic arrive + sleep-spin.
    __threadfence();
    __syncthreads();
    if (tid == 0) {
      atomicAdd(&cnt[t], 1u);
      volatile unsigned* c = &cnt[t];
      while (*c < (unsigned)NBLK) __builtin_amdgcn_s_sleep(1);
    }
    __syncthreads();
    __threadfence();                             // acquire before next h load
  }
}

// ------------------------------ launcher -----------------------------------
extern "C" void kernel_launch(void* const* d_in, const int* in_sizes, int n_in,
                              void* d_out, int out_size, void* d_ws, size_t ws_size,
                              hipStream_t stream) {
  (void)in_sizes; (void)n_in; (void)out_size; (void)ws_size;
  const float* x     = (const float*)d_in[0];
  const float* W_in  = (const float*)d_in[1];
  const float* b_in  = (const float*)d_in[2];
  const float* W_hh  = (const float*)d_in[3];
  const float* b_hh  = (const float*)d_in[4];
  const float* noise = (const float*)d_in[5];
  float* out = (float*)d_out;

  // Workspace: [0,16KB) step counters (4096 u32); [16KB,48KB) bf16 h buffer.
  unsigned*       cnt  = (unsigned*)d_ws;
  unsigned short* hbuf = (unsigned short*)((char*)d_ws + 16384);

  const int zero_words = SEQ + (BATCH * HID * 2) / 4;   // 12288 u32
  rnn_zero_ws_kernel<<<(zero_words + 255) / 256, 256, 0, stream>>>(cnt, zero_words);

  dim3 g1(SEQ * BATCH / 16, HID / 128);
  rnn_proj_kernel<<<g1, 256, 0, stream>>>(x, W_in, b_in, noise, out);

  rnn_scan_kernel<<<NBLK, 128, 0, stream>>>(W_hh, b_hh, out, cnt, hbuf);
}
